// SimplifiedNodeRankingGNN_65695819759677
// MI455X (gfx1250) — matrix operand order repaired and checked by reading, not compiled, not run on hardware
//
#include <hip/hip_runtime.h>

// ---------------- problem constants (from reference) ----------------
#define Nn    50000
#define Ee    800000
#define ETOT  850000      // E + N self loops
#define IN_   256
#define HID_  128
#define NHEAD 4
#define MT    5           // row tiles per wave (50000 = 625 * 80)
#define ROWS  80          // MT * 16
#define APAD  8           // LDS row pad (ushorts) -> 16B, 4-bank lane skew

typedef __attribute__((ext_vector_type(16))) __bf16 v16bf;
typedef __attribute__((ext_vector_type(8)))  float  v8f;

union V16U { v16bf v; unsigned short s[16]; };

// ---------------- small device helpers ----------------
__device__ __forceinline__ unsigned short f2bf(float f) {
  unsigned u = __float_as_uint(f);
  return (unsigned short)((u + 0x7fffu + ((u >> 16) & 1u)) >> 16); // RNE
}
// order-preserving float<->u32 map for atomicMax-based segment max
__device__ __forceinline__ unsigned flip_f(float f) {
  unsigned u = __float_as_uint(f);
  return (u & 0x80000000u) ? ~u : (u | 0x80000000u);
}
__device__ __forceinline__ float unflip_f(unsigned u) {
  return __uint_as_float((u & 0x80000000u) ? (u ^ 0x80000000u) : ~u);
}
__device__ __forceinline__ float wave_sum(float p) {
  #pragma unroll
  for (int o = 16; o > 0; o >>= 1) p += __shfl_xor(p, o, 32);
  return p;
}

// ---------------- fp32 -> bf16 conversion (row-major, for activations) ----------
__global__ void cvt_f32_bf16(const float* __restrict__ in,
                             unsigned short* __restrict__ out, long long n) {
  long long i  = (long long)blockIdx.x * blockDim.x + threadIdx.x;
  long long st = (long long)gridDim.x * blockDim.x;
  for (; i < n; i += st) out[i] = f2bf(in[i]);
}

// ---------------- weight pack: fp32 [K x Nc] -> bf16 WMMA B-fragment layout -----
// out[((kt*(Nc/16)+nt)*32 + lane)*16 + e] = W[(kt*32 + (lane>>4)*16 + e)*Nc + nt*16 + (lane&15)]
// => in the GEMM each lane's 16 B elements are 32 contiguous bytes (2x b128 loads)
__global__ void pack_w(const float* __restrict__ W, unsigned short* __restrict__ out,
                       int K, int Nc) {
  int n = K * Nc, tiles_n = Nc >> 4;
  int i  = blockIdx.x * blockDim.x + threadIdx.x;
  int st = gridDim.x * blockDim.x;
  for (; i < n; i += st) {
    int e    = i & 15;
    int lane = (i >> 4) & 31;
    int rest = i >> 9;
    int nt   = rest % tiles_n;
    int kt   = rest / tiles_n;
    int k    = kt * 32 + (lane >> 4) * 16 + e;
    int c    = nt * 16 + (lane & 15);
    out[i] = f2bf(W[(size_t)k * Nc + c]);
  }
}

// ---------------- per-layer scratch init ----------------
__global__ void init_layer(float* __restrict__ accum, unsigned* __restrict__ amax,
                           float* __restrict__ den, long long nAcc, long long nHd) {
  long long i  = (long long)blockIdx.x * blockDim.x + threadIdx.x;
  long long st = (long long)gridDim.x * blockDim.x;
  for (; i < nAcc; i += st) {
    accum[i] = 0.f;
    if (i < nHd) { amax[i] = 0x007FFFFFu; /* flip(-inf) */ den[i] = 0.f; }
  }
}

// ---------------- WMMA bf16 GEMM: C[MxNc] = A[MxK] @ B[KxNc] + bias (opt. ELU) --
// grid.x = M/80; blockDim.x = 32*(Nc/16); wave w -> cols [16w,16w+16), 5 row tiles.
// A panel (80 x K) staged in LDS (padded rows), B pre-packed per-fragment.
__global__ void wmma_gemm_bias(const unsigned short* __restrict__ A,
                               const unsigned short* __restrict__ Bp,
                               const float* __restrict__ bias,
                               float* __restrict__ C,
                               int M, int K, int Nc, int act) {
  __shared__ __align__(16) unsigned short As[ROWS * (IN_ + APAD)]; // max K=256
  const int stride  = K + APAD;
  const int tiles_n = Nc >> 4;
  const int wave = threadIdx.x >> 5;
  const int lane = threadIdx.x & 31;
  const int half = lane >> 4;
  const int l16  = lane & 15;
  const int row0 = blockIdx.x * ROWS;

  // cooperative copy: 80 contiguous rows of A -> padded LDS rows, 16B chunks
  {
    const unsigned short* Ag = A + (size_t)row0 * K;
    const int chunks = ROWS * (K >> 3);          // 8-ushort chunks
    for (int v = threadIdx.x; v < chunks; v += blockDim.x) {
      int r = v / (K >> 3), c = v - r * (K >> 3);
      *(uint4*)(&As[r * stride + c * 8]) = *(const uint4*)(Ag + (size_t)r * K + c * 8);
    }
  }
  __syncthreads();

  v8f acc[MT] = {};
  for (int k0 = 0; k0 < K; k0 += 32) {
    V16U b;
    // packed B fragment: 32 contiguous bytes per lane
    const unsigned short* Bf = Bp + ((size_t)((k0 >> 5) * tiles_n + wave) * 32 + lane) * 16;
    #pragma unroll
    for (int e = 0; e < 16; ++e) b.s[e] = Bf[e];
    #pragma unroll
    for (int t = 0; t < MT; ++t) {
      V16U a;
      // A 16x32 fragment from LDS: half0 -> K 0..15, half1 -> K 16..31 (matches packed B)
      const unsigned short* Ar = &As[(t * 16 + l16) * stride + k0 + half * 16];
      #pragma unroll
      for (int e = 0; e < 16; ++e) { a.s[e] = Ar[e]; }
      acc[t] = __builtin_amdgcn_wmma_f32_16x16x32_bf16(false, a.v, false, b.v,
                                                       (short)0, acc[t], false, false);
    }
  }

  const int   c  = (wave << 4) + l16;
  const float bi = bias[c];
  #pragma unroll
  for (int t = 0; t < MT; ++t) {
    #pragma unroll
    for (int v = 0; v < 8; ++v) {
      int r = row0 + t * 16 + half * 8 + v;      // C/D layout: lanes 16-31 -> M+8
      float val = acc[t][v] + bi;
      if (act) val = val > 0.f ? val : (__expf(val) - 1.f);   // ELU
      C[(size_t)r * Nc + c] = val;
    }
  }
}

// ---------------- pass 1: per-edge attention logits + segment max ----------------
__global__ void edge_alpha(const float* __restrict__ xl, const float* __restrict__ xr,
                           const long long* __restrict__ ei, const float* __restrict__ att,
                           float* __restrict__ alpha, unsigned* __restrict__ amax) {
  int e = (int)(((long long)blockIdx.x * blockDim.x + threadIdx.x) >> 5);
  if (e >= ETOT) return;
  int lane = threadIdx.x & 31;
  long long s, d;
  if (e < Ee) { s = ei[e]; d = ei[Ee + e]; } else { s = d = (long long)(e - Ee); }
  const float* xs = xl + (size_t)s * HID_;
  const float* xd = xr + (size_t)d * HID_;
  #pragma unroll
  for (int h = 0; h < NHEAD; ++h) {
    float v = xs[h * 32 + lane] + xd[h * 32 + lane];
    v = v > 0.f ? v : 0.2f * v;                 // leaky_relu(0.2)
    float p = wave_sum(v * att[h * 32 + lane]); // dot over DH=32
    if (lane == 0) {
      alpha[(size_t)e * NHEAD + h] = p;
      atomicMax(&amax[(size_t)d * NHEAD + h], flip_f(p));
    }
  }
}

// ---------------- pass 2: exp, segment sums (den + ex-weighted features) --------
__global__ void edge_scatter(const float* __restrict__ xl, const long long* __restrict__ ei,
                             const float* __restrict__ alpha, const unsigned* __restrict__ amax,
                             float* __restrict__ den, float* __restrict__ accum) {
  int e = (int)(((long long)blockIdx.x * blockDim.x + threadIdx.x) >> 5);
  if (e >= ETOT) return;
  int lane = threadIdx.x & 31;
  long long s, d;
  if (e < Ee) { s = ei[e]; d = ei[Ee + e]; } else { s = d = (long long)(e - Ee); }
  const float* xs  = xl + (size_t)s * HID_;
  float*       acd = accum + (size_t)d * HID_;
  #pragma unroll
  for (int h = 0; h < NHEAD; ++h) {
    float ex = __expf(alpha[(size_t)e * NHEAD + h] -
                      unflip_f(amax[(size_t)d * NHEAD + h]));
    if (lane == 0) atomicAdd(&den[(size_t)d * NHEAD + h], ex);
    atomicAdd(&acd[h * 32 + lane], ex * xs[h * 32 + lane]);
  }
}

// ---------------- per-node: normalize + bias + residual + LayerNorm (+ELU) ------
__global__ void node_finalize(const float* __restrict__ accum, const float* __restrict__ den,
                              const float* __restrict__ cbias, const float* __restrict__ hres,
                              const float* __restrict__ g, const float* __restrict__ bln,
                              float* __restrict__ hout, unsigned short* __restrict__ hbf,
                              int do_elu) {
  int node = (int)(((long long)blockIdx.x * blockDim.x + threadIdx.x) >> 5);
  if (node >= Nn) return;
  int lane = threadIdx.x & 31;
  const float* ac = accum + (size_t)node * HID_;
  const float* hr = hres  + (size_t)node * HID_;
  float vals[4], s = 0.f;
  #pragma unroll
  for (int j = 0; j < 4; ++j) {                   // j == head index
    int f = j * 32 + lane;
    float dn = den[(size_t)node * NHEAD + j] + 1e-16f;
    float v  = ac[f] / dn + cbias[f] + hr[f];     // gat out + bias + residual
    vals[j] = v; s += v;
  }
  float mean = wave_sum(s) * (1.f / 128.f);
  float vs = 0.f;
  #pragma unroll
  for (int j = 0; j < 4; ++j) { float t = vals[j] - mean; vs += t * t; }
  float inv = rsqrtf(wave_sum(vs) * (1.f / 128.f) + 1e-5f);
  #pragma unroll
  for (int j = 0; j < 4; ++j) {
    int f = j * 32 + lane;
    float y = (vals[j] - mean) * inv * g[f] + bln[f];
    if (do_elu) y = y > 0.f ? y : (__expf(y) - 1.f);
    hout[(size_t)node * HID_ + f] = y;
    hbf [(size_t)node * HID_ + f] = f2bf(y);
  }
}

// ---------------- scorer: out[n] = elu(h@s1)@s2 + s2_b (s1 part done by GEMM) ---
__global__ void score_kernel(const float* __restrict__ s1out, const float* __restrict__ s2W,
                             const float* __restrict__ s2b, float* __restrict__ out) {
  int node = (int)(((long long)blockIdx.x * blockDim.x + threadIdx.x) >> 5);
  if (node >= Nn) return;
  int lane = threadIdx.x & 31;
  const float* r = s1out + (size_t)node * 64;
  float p = r[lane] * s2W[lane] + r[lane + 32] * s2W[lane + 32];
  p = wave_sum(p);
  if (lane == 0) out[node] = p + s2b[0];
}

// ---------------- host side ----------------
extern "C" void kernel_launch(void* const* d_in, const int* in_sizes, int n_in,
                              void* d_out, int out_size, void* d_ws, size_t ws_size,
                              hipStream_t stream) {
  const float*     x        = (const float*)d_in[0];
  const long long* ei       = (const long long*)d_in[1];   // int64 edge_index [2,E]
  const float*     inW      = (const float*)d_in[2];
  const float*     inB      = (const float*)d_in[3];
  const float*     Wl       = (const float*)d_in[4];
  const float*     bl       = (const float*)d_in[5];
  const float*     Wr       = (const float*)d_in[6];
  const float*     br       = (const float*)d_in[7];
  const float*     att      = (const float*)d_in[8];
  const float*     cbias    = (const float*)d_in[9];
  const float*     lng      = (const float*)d_in[10];
  const float*     lnb      = (const float*)d_in[11];
  const float*     s1W      = (const float*)d_in[12];
  const float*     s1b      = (const float*)d_in[13];
  const float*     s2W      = (const float*)d_in[14];
  const float*     s2b      = (const float*)d_in[15];

  char* base = (char*)d_ws;
  const size_t OFF_XBF   = 0;                       // 50000*256*2  = 25.6 MB
  const size_t OFF_H     = 25600000;                // h fp32       = 25.6 MB
  const size_t OFF_HBF   = 51200000;                // h bf16       = 12.8 MB
  const size_t OFF_XL    = 64000000;                // xl fp32 (reused for s1out)
  const size_t OFF_XR    = 89600000;                // xr fp32
  const size_t OFF_ALPHA = 115200000;               // 850000*4*4   = 13.6 MB
  const size_t OFF_AMAX  = 128800000;               // 50000*4*4
  const size_t OFF_DEN   = 129600000;
  const size_t OFF_WBF   = 130400000;               // packed bf16 weights

  unsigned short* xbf   = (unsigned short*)(base + OFF_XBF);
  float*          accum = (float*)(base + OFF_XBF); // reuse: x_bf16 dead after in_proj
  float*          hA    = (float*)(base + OFF_H);
  unsigned short* hbf   = (unsigned short*)(base + OFF_HBF);
  float*          xl    = (float*)(base + OFF_XL);
  float*          s1out = (float*)(base + OFF_XL);  // reuse after last layer
  float*          xr    = (float*)(base + OFF_XR);
  float*          alpha = (float*)(base + OFF_ALPHA);
  unsigned*       amax  = (unsigned*)(base + OFF_AMAX);
  float*          den   = (float*)(base + OFF_DEN);
  unsigned short* wbf   = (unsigned short*)(base + OFF_WBF);
  unsigned short* wInP  = wbf;                      // 256*128 (packed)
  unsigned short* wL    = wbf + 32768;              // 3*128*128
  unsigned short* wR    = wbf + 32768 + 49152;      // 3*128*128
  unsigned short* wS1   = wbf + 131072;             // 128*64

  // ---- activation conversion + weight packing ----
  cvt_f32_bf16<<<2048, 256, 0, stream>>>(x, xbf, (long long)Nn * IN_);
  pack_w<<<64, 256, 0, stream>>>(inW, wInP, IN_, HID_);
  for (int i = 0; i < 3; ++i) {
    pack_w<<<32, 256, 0, stream>>>(Wl + i * 16384, wL + i * 16384, HID_, HID_);
    pack_w<<<32, 256, 0, stream>>>(Wr + i * 16384, wR + i * 16384, HID_, HID_);
  }
  pack_w<<<16, 256, 0, stream>>>(s1W, wS1, HID_, 64);

  // ---- h = x @ in_proj_W + b ----
  wmma_gemm_bias<<<Nn / ROWS, 256, 0, stream>>>(xbf, wInP, inB, hA, Nn, IN_, HID_, 0);
  cvt_f32_bf16<<<2048, 256, 0, stream>>>(hA, hbf, (long long)Nn * HID_);

  const int edgeBlocks = (ETOT * 32) / 256;         // 106250, exact
  for (int i = 0; i < 3; ++i) {
    wmma_gemm_bias<<<Nn / ROWS, 256, 0, stream>>>(hbf, wL + i * 16384, bl + i * HID_,
                                                  xl, Nn, HID_, HID_, 0);
    wmma_gemm_bias<<<Nn / ROWS, 256, 0, stream>>>(hbf, wR + i * 16384, br + i * HID_,
                                                  xr, Nn, HID_, HID_, 0);
    init_layer<<<4096, 256, 0, stream>>>(accum, amax, den,
                                         (long long)Nn * HID_, (long long)Nn * NHEAD);
    edge_alpha<<<edgeBlocks, 256, 0, stream>>>(xl, xr, ei, att + i * HID_, alpha, amax);
    edge_scatter<<<edgeBlocks, 256, 0, stream>>>(xl, ei, alpha, amax, den, accum);
    float* hout = (i == 2) ? ((float*)d_out + Nn) : hA;   // final h -> d_out[N:]
    node_finalize<<<Nn / 8, 256, 0, stream>>>(accum, den, cbias + i * HID_, hA,
                                              lng + i * HID_, lnb + i * HID_,
                                              hout, hbf, (i < 2) ? 1 : 0);
  }

  // ---- scores = elu(h @ s1 + b1) @ s2 + b2 ----
  wmma_gemm_bias<<<Nn / ROWS, 128, 0, stream>>>(hbf, wS1, s1b, s1out, Nn, HID_, 64, 1);
  score_kernel<<<Nn / 8, 256, 0, stream>>>(s1out, s2W, s2b, (float*)d_out);
}
